// TestNet55_desc_23055384445043
// MI455X (gfx1250) — compile-verified, matrix operand order checked by reading, not run on hardware
//
#include <hip/hip_runtime.h>

typedef __attribute__((ext_vector_type(16))) _Float16 v16h;
typedef __attribute__((ext_vector_type(8)))  _Float16 v8h;
typedef __attribute__((ext_vector_type(4)))  _Float16 v4h;
typedef __attribute__((ext_vector_type(8)))  float    v8f;

#define NNODES 100000
#define NEDGES 600000
#define NGRAPH 16

// ---------------- degree / normalization ----------------
__global__ void deg_count_kernel(const int* __restrict__ dst, float* __restrict__ deg, int E) {
    int e = blockIdx.x * blockDim.x + threadIdx.x;
    if (e < E) atomicAdd(&deg[dst[e]], 1.0f);
}

__global__ void dinv_kernel(float* __restrict__ deg, int n) {
    int i = blockIdx.x * blockDim.x + threadIdx.x;
    if (i < n) deg[i] = rsqrtf(deg[i] + 1.0f);   // +1 self-loop => deg >= 1
}

// ---------------- input prep: pos f32 [N,3] -> f16 [N,32] zero-padded ----------------
__global__ void prep_pos_kernel(const float* __restrict__ pos, _Float16* __restrict__ ph, int total) {
    int idx = blockIdx.x * blockDim.x + threadIdx.x;
    if (idx >= total) return;
    int i = idx >> 5, c = idx & 31;
    ph[idx] = (_Float16)((c < 3) ? pos[i * 3 + c] : 0.0f);
}

// ---------------- weight prep: f32 [CINV,COUTV] row-major -> f16 WMMA-fragment order ----------------
// Wf[(((ct*KSTEPS)+ks)*32 + lane)*16 + e], zero-padded to CINP x COUTP.
template<int CINP, int COUTP, int CINV, int COUTV>
__global__ void prep_w_kernel(const float* __restrict__ W, _Float16* __restrict__ Wf) {
    constexpr int KSTEPS = CINP / 32;
    int idx = blockIdx.x * blockDim.x + threadIdx.x;
    if (idx >= CINP * COUTP) return;
    int e    = idx & 15;
    int lane = (idx >> 4) & 31;
    int frag = idx >> 9;
    int ct = frag / KSTEPS, ks = frag % KSTEPS;
    int col = ct * 16 + (lane & 15);
    int kg  = (lane >> 4) << 3;
    int k   = ks * 32 + ((e < 8) ? (kg + e) : (16 + kg + (e - 8)));
    float v = (k < CINV && col < COUTV) ? W[k * COUTV + col] : 0.0f;
    Wf[idx] = (_Float16)v;
}

// ---------------- WMMA GEMM, NT column tiles per wave (A-fragment register reuse) ----------------
template<int CIN, int COUT, int NT, int NVALID, bool BIAS, bool RELU, bool HALFOUT>
__global__ void gemm_wmma_kernel(const _Float16* __restrict__ X, const _Float16* __restrict__ Wf,
                                 const float* __restrict__ bias, void* __restrict__ Yv,
                                 int nrows, int ldY)
{
    constexpr int KSTEPS = CIN / 32;
    constexpr int COLT   = COUT / 16;
    static_assert(COLT % NT == 0, "NT must divide COLT");
    constexpr int CGRP   = COLT / NT;
    const int lane = threadIdx.x & 31;
    const int wave = (blockIdx.x * blockDim.x + threadIdx.x) >> 5;
    const int rt  = wave / CGRP;
    const int ct0 = (wave % CGRP) * NT;
    if (rt * 16 >= nrows) return;            // wave-uniform: EXEC stays all-1s for WMMA

    const int kg   = (lane >> 4) << 3;       // 0 or 8
    const int arow = rt * 16 + (lane & 15);
    const _Float16* xr = X + (size_t)arow * CIN;
    const _Float16* wr = Wf + ((size_t)ct0 * KSTEPS * 32 + lane) * 16;

    v8f acc[NT];
    #pragma unroll
    for (int t = 0; t < NT; ++t) acc[t] = (v8f){};

    #pragma unroll
    for (int ks = 0; ks < KSTEPS; ++ks) {
        v8h alo = *(const v8h*)(xr + ks * 32 + kg);          // 16B vector load
        v8h ahi = *(const v8h*)(xr + ks * 32 + 16 + kg);
        v16h a;
        #pragma unroll
        for (int e = 0; e < 8; ++e) { a[e] = alo[e]; a[8 + e] = ahi[e]; }
        #pragma unroll
        for (int t = 0; t < NT; ++t) {
            v16h b = *(const v16h*)(wr + ((size_t)t * KSTEPS + ks) * 32 * 16); // 32B fragment
            acc[t] = __builtin_amdgcn_wmma_f32_16x16x32_f16(false, a, false, b, (short)0, acc[t],
                                                            false, false);
        }
    }

    const int mb = (lane >> 4) << 3;
    #pragma unroll
    for (int t = 0; t < NT; ++t) {
        const int bcol = (ct0 + t) * 16 + (lane & 15);
        const bool ok = (NVALID == COUT) || (bcol < NVALID);
        float bb = 0.0f;
        if constexpr (BIAS) { if (ok) bb = bias[bcol]; }
        #pragma unroll
        for (int r = 0; r < 8; ++r) {
            int row = rt * 16 + mb + r;
            float v = acc[t][r];
            if constexpr (BIAS) v += bb;
            if constexpr (RELU) v = fmaxf(v, 0.0f);
            if (ok) {
                if constexpr (HALFOUT) ((_Float16*)Yv)[(size_t)row * ldY + bcol] = (_Float16)v;
                else                   ((float*)Yv)[(size_t)row * ldY + bcol] = v;
            }
        }
    }
}

// ---------------- GCN aggregation (float4-vectorized) ----------------
__global__ void selfloop_kernel(const float* __restrict__ xw, const float* __restrict__ dinv,
                                float* __restrict__ agg, int C4, int total4) {
    int idx = blockIdx.x * blockDim.x + threadIdx.x;
    if (idx >= total4) return;
    int i = idx / C4;
    float d = dinv[i];
    float w = d * d;
    float4 v = *(const float4*)(xw + (size_t)idx * 4);
    float4 o = make_float4(w * v.x, w * v.y, w * v.z, w * v.w);
    *(float4*)(agg + (size_t)idx * 4) = o;
}

__global__ void scatter_kernel(const int* __restrict__ src, const int* __restrict__ dst,
                               const float* __restrict__ dinv, const float* __restrict__ xw,
                               float* __restrict__ agg, int C4, int total4) {
    int idx = blockIdx.x * blockDim.x + threadIdx.x;
    if (idx >= total4) return;
    int e = idx / C4, q = idx % C4;
    int s = src[e], d = dst[e];
    float w = dinv[s] * dinv[d];
    float4 v = *(const float4*)(xw + ((size_t)s * C4 + q) * 4);
    float* out = agg + ((size_t)d * C4 + q) * 4;
    atomicAdd(out + 0, w * v.x);
    atomicAdd(out + 1, w * v.y);
    atomicAdd(out + 2, w * v.z);
    atomicAdd(out + 3, w * v.w);
}

// ---------------- BatchNorm (training mode, biased var) ----------------
__global__ void bn_stats_kernel(const float* __restrict__ x, float* __restrict__ stats,
                                int C4, int total4) {
    __shared__ float s_sum[128];
    __shared__ float s_sq[128];
    for (int c = threadIdx.x; c < C4 * 4; c += blockDim.x) { s_sum[c] = 0.0f; s_sq[c] = 0.0f; }
    __syncthreads();
    for (int idx = blockIdx.x * blockDim.x + threadIdx.x; idx < total4; idx += gridDim.x * blockDim.x) {
        float4 v = *(const float4*)(x + (size_t)idx * 4);
        int c = (idx % C4) * 4;
        atomicAdd(&s_sum[c + 0], v.x); atomicAdd(&s_sq[c + 0], v.x * v.x);
        atomicAdd(&s_sum[c + 1], v.y); atomicAdd(&s_sq[c + 1], v.y * v.y);
        atomicAdd(&s_sum[c + 2], v.z); atomicAdd(&s_sq[c + 2], v.z * v.z);
        atomicAdd(&s_sum[c + 3], v.w); atomicAdd(&s_sq[c + 3], v.w * v.w);
    }
    __syncthreads();
    for (int c = threadIdx.x; c < C4 * 4; c += blockDim.x) {
        atomicAdd(&stats[c], s_sum[c]);
        atomicAdd(&stats[128 + c], s_sq[c]);
    }
}

__global__ void bn_finalize_kernel(const float* __restrict__ stats, const float* __restrict__ gamma,
                                   const float* __restrict__ beta, float* __restrict__ scale,
                                   float* __restrict__ shift, int Cvalid, float invN) {
    int c = blockIdx.x * blockDim.x + threadIdx.x;
    if (c >= Cvalid) return;
    float m = stats[c] * invN;
    float v = stats[128 + c] * invN - m * m;
    float sc = rsqrtf(v + 1e-5f) * gamma[c];
    scale[c] = sc;
    shift[c] = beta[c] - m * sc;
}

// fused: BN affine + ReLU + f32->f16 (zero pad channels) for next layer's A matrix
__global__ void bn_apply_relu_h_kernel(const float* __restrict__ x, const float* __restrict__ scale,
                                       const float* __restrict__ shift, _Float16* __restrict__ out,
                                       int C4, int Cvalid, int total4) {
    int idx = blockIdx.x * blockDim.x + threadIdx.x;
    if (idx >= total4) return;
    int c = (idx % C4) * 4;
    float4 v = *(const float4*)(x + (size_t)idx * 4);
    float r0 = (c + 0 < Cvalid) ? fmaxf(v.x * scale[c + 0] + shift[c + 0], 0.0f) : 0.0f;
    float r1 = (c + 1 < Cvalid) ? fmaxf(v.y * scale[c + 1] + shift[c + 1], 0.0f) : 0.0f;
    float r2 = (c + 2 < Cvalid) ? fmaxf(v.z * scale[c + 2] + shift[c + 2], 0.0f) : 0.0f;
    float r3 = (c + 3 < Cvalid) ? fmaxf(v.w * scale[c + 3] + shift[c + 3], 0.0f) : 0.0f;
    v4h h; h[0] = (_Float16)r0; h[1] = (_Float16)r1; h[2] = (_Float16)r2; h[3] = (_Float16)r3;
    *(v4h*)(out + (size_t)idx * 4) = h;
}

// ---------------- global sum pool + convert ----------------
__global__ void pool_kernel(const float* __restrict__ x, const int* __restrict__ batch,
                            float* __restrict__ pool, int C4, int total4) {
    int idx = blockIdx.x * blockDim.x + threadIdx.x;
    if (idx >= total4) return;
    int i = idx / C4, q = idx % C4;
    float4 v = *(const float4*)(x + (size_t)idx * 4);
    float* out = pool + ((size_t)batch[i] * C4 + q) * 4;
    atomicAdd(out + 0, v.x);
    atomicAdd(out + 1, v.y);
    atomicAdd(out + 2, v.z);
    atomicAdd(out + 3, v.w);
}

__global__ void f32_to_f16_kernel(const float* __restrict__ in, _Float16* __restrict__ out, int n) {
    int idx = blockIdx.x * blockDim.x + threadIdx.x;
    if (idx < n) out[idx] = (_Float16)in[idx];
}

extern "C" void kernel_launch(void* const* d_in, const int* in_sizes, int n_in,
                              void* d_out, int out_size, void* d_ws, size_t ws_size,
                              hipStream_t stream) {
    (void)in_sizes; (void)n_in; (void)out_size; (void)ws_size;

    const float* pos  = (const float*)d_in[0];
    const int*   edge = (const int*)d_in[1];          // [2,E]: src = edge, dst = edge+E
    const int*   batch= (const int*)d_in[2];
    const float* w1   = (const float*)d_in[3];        // b1/b2/b3 cancel in training-mode BN
    const float* g1   = (const float*)d_in[5];
    const float* be1  = (const float*)d_in[6];
    const float* w2   = (const float*)d_in[7];
    const float* g2   = (const float*)d_in[9];
    const float* be2  = (const float*)d_in[10];
    const float* w3   = (const float*)d_in[11];
    const float* g3   = (const float*)d_in[13];
    const float* be3  = (const float*)d_in[14];
    const float* fw0  = (const float*)d_in[15];
    const float* fb0  = (const float*)d_in[16];
    const float* fw1  = (const float*)d_in[17];
    const float* fb1  = (const float*)d_in[18];
    const float* fw2  = (const float*)d_in[19];
    const float* fb2  = (const float*)d_in[20];
    const float* fw3  = (const float*)d_in[21];
    const float* fb3  = (const float*)d_in[22];

    // ---- workspace layout (~135 MB total, L2-resident on MI455X) ----
    float* ws    = (float*)d_ws;
    float* dinv  = ws;                                    // N f32
    float* A     = dinv + NNODES;                         // N*128 f32 (GEMM out / xw)
    float* B     = A + (size_t)NNODES * 128;              // N*128 f32 (aggregation)
    float* stats = B + (size_t)NNODES * 128;              // 256
    float* scale = stats + 256;                           // 128
    float* shift = scale + 128;                           // 128
    float* pool  = shift + 128;                           // 2048
    _Float16* H    = (_Float16*)(pool + 2048);            // N*128 f16 activations
    _Float16* posh = H + (size_t)NNODES * 128;            // N*32 f16
    _Float16* poolh= posh + (size_t)NNODES * 32;          // 2048
    _Float16* s1h  = poolh + 2048;                        // 2048
    _Float16* s2h  = s1h + 2048;                          // 2048
    _Float16* wf1  = s2h + 2048;                          // 32*64
    _Float16* wf2  = wf1 + 32 * 64;                       // 64*96
    _Float16* wf3  = wf2 + 64 * 96;                       // 96*128
    _Float16* wff0 = wf3 + 96 * 128;                      // 128*128
    _Float16* wff1 = wff0 + 128 * 128;                    // 128*128
    _Float16* wff2 = wff1 + 128 * 128;                    // 128*128
    _Float16* wff3 = wff2 + 128 * 128;                    // 128*112

    // ---- structural normalization (computed once, shared by all 3 conv layers) ----
    hipMemsetAsync(dinv, 0, NNODES * sizeof(float), stream);
    deg_count_kernel<<<(NEDGES + 255) / 256, 256, 0, stream>>>(edge + NEDGES, dinv, NEDGES);
    dinv_kernel<<<(NNODES + 255) / 256, 256, 0, stream>>>(dinv, NNODES);

    // ---- weight + input prep (tiny) ----
    prep_pos_kernel<<<((NNODES * 32) + 255) / 256, 256, 0, stream>>>(pos, posh, NNODES * 32);
    prep_w_kernel< 32,  64,   3,  64><<<(32 * 64 + 255) / 256, 256, 0, stream>>>(w1,  wf1);
    prep_w_kernel< 64,  96,  64,  94><<<(64 * 96 + 255) / 256, 256, 0, stream>>>(w2,  wf2);
    prep_w_kernel< 96, 128,  94, 128><<<(96 * 128 + 255) / 256, 256, 0, stream>>>(w3,  wf3);
    prep_w_kernel<128, 128, 128, 128><<<(128 * 128 + 255) / 256, 256, 0, stream>>>(fw0, wff0);
    prep_w_kernel<128, 128, 128, 128><<<(128 * 128 + 255) / 256, 256, 0, stream>>>(fw1, wff1);
    prep_w_kernel<128, 128, 128, 128><<<(128 * 128 + 255) / 256, 256, 0, stream>>>(fw2, wff2);
    prep_w_kernel<128, 112, 128, 100><<<(128 * 112 + 255) / 256, 256, 0, stream>>>(fw3, wff3);

    #define GEMM_BLOCKS(nrows, COUT, NT) ((((nrows + 15) / 16) * ((COUT / 16) / NT) + 7) / 8)

    // ---- GCN aggregation + BN helper ----
    auto agg_bn = [&](int Cpad, int Cvalid, const float* gamma, const float* beta) {
        int C4 = Cpad / 4;
        int totalN4 = NNODES * C4;
        int totalE4 = NEDGES * C4;
        selfloop_kernel<<<(totalN4 + 255) / 256, 256, 0, stream>>>(A, dinv, B, C4, totalN4);
        scatter_kernel<<<(totalE4 + 255) / 256, 256, 0, stream>>>(edge, edge + NEDGES, dinv, A, B, C4, totalE4);
        hipMemsetAsync(stats, 0, 256 * sizeof(float), stream);
        bn_stats_kernel<<<1024, 256, 0, stream>>>(B, stats, C4, totalN4);
        bn_finalize_kernel<<<1, 128, 0, stream>>>(stats, gamma, beta, scale, shift, Cvalid, 1.0f / NNODES);
        bn_apply_relu_h_kernel<<<(totalN4 + 255) / 256, 256, 0, stream>>>(B, scale, shift, H, C4, Cvalid, totalN4);
    };

    // ---- layer 1: [N,32]@[32,64], NT=4 (whole row strip per wave) ----
    gemm_wmma_kernel<32, 64, 4, 64, false, false, false>
        <<<GEMM_BLOCKS(NNODES, 64, 4), 256, 0, stream>>>(posh, wf1, nullptr, A, NNODES, 64);
    agg_bn(64, 64, g1, be1);

    // ---- layer 2: [N,64]@[64,96] (94 valid), NT=3 ----
    gemm_wmma_kernel<64, 96, 3, 96, false, false, false>
        <<<GEMM_BLOCKS(NNODES, 96, 3), 256, 0, stream>>>(H, wf2, nullptr, A, NNODES, 96);
    agg_bn(96, 94, g2, be2);

    // ---- layer 3: [N,96]@[96,128], NT=4 ----
    gemm_wmma_kernel<96, 128, 4, 128, false, false, false>
        <<<GEMM_BLOCKS(NNODES, 128, 4), 256, 0, stream>>>(H, wf3, nullptr, A, NNODES, 128);
    agg_bn(128, 128, g3, be3);

    // ---- fw0 + relu, NT=4 ----
    gemm_wmma_kernel<128, 128, 4, 128, true, true, false>
        <<<GEMM_BLOCKS(NNODES, 128, 4), 256, 0, stream>>>(H, wff0, fb0, A, NNODES, 128);

    // ---- global sum pool ----
    hipMemsetAsync(pool, 0, NGRAPH * 128 * sizeof(float), stream);
    pool_kernel<<<((NNODES * 32) + 255) / 256, 256, 0, stream>>>(A, batch, pool, 32, NNODES * 32);
    f32_to_f16_kernel<<<8, 256, 0, stream>>>(pool, poolh, NGRAPH * 128);

    // ---- MLP head on [16,128] ----
    gemm_wmma_kernel<128, 128, 1, 128, true, true, true>
        <<<GEMM_BLOCKS(NGRAPH, 128, 1), 256, 0, stream>>>(poolh, wff1, fb1, s1h, NGRAPH, 128);
    gemm_wmma_kernel<128, 128, 1, 128, true, true, true>
        <<<GEMM_BLOCKS(NGRAPH, 128, 1), 256, 0, stream>>>(s1h, wff2, fb2, s2h, NGRAPH, 128);
    gemm_wmma_kernel<128, 112, 1, 100, true, false, false>
        <<<GEMM_BLOCKS(NGRAPH, 112, 1), 256, 0, stream>>>(s2h, wff3, fb3, d_out, NGRAPH, 100);

    #undef GEMM_BLOCKS
}
